// RecyclingAllFeatures_43791486550462
// MI455X (gfx1250) — matrix operand order repaired and checked by reading, not compiled, not run on hardware
//
#include <hip/hip_runtime.h>
#include <hip/hip_bf16.h>

typedef __attribute__((ext_vector_type(16))) _Float16 v16h;
typedef __attribute__((ext_vector_type(8)))  float    v8f;
typedef __attribute__((ext_vector_type(4)))  float    f4;

#define LSEQ   1024
#define DPAIR  128
#define DSTATE 32
#define DMSA   256
#define DRBF   64

// ---------------------------------------------------------------------------
// Kernel 1: state layernorm + precompute sl = LN(state)@Wl + b_dist,
//           sr = LN(state)@Wr.  One block (128 thr) per residue.
// ---------------------------------------------------------------------------
__global__ void __launch_bounds__(128) state_prep_kernel(
    const float* __restrict__ state,
    const float* __restrict__ g_state, const float* __restrict__ be_state,
    const float* __restrict__ W_dist,  const float* __restrict__ b_dist,
    float* __restrict__ state_n_out,
    float* __restrict__ sl, float* __restrict__ sr)
{
    __shared__ float sn[DSTATE];
    const int i = blockIdx.x, tid = threadIdx.x;
    if (tid < DSTATE) {
        float x = state[i * DSTATE + tid];
        float s = x, s2 = x * x;
        #pragma unroll
        for (int off = 16; off; off >>= 1) {
            s  += __shfl_xor(s,  off, 32);
            s2 += __shfl_xor(s2, off, 32);
        }
        float mu  = s  * (1.f / DSTATE);
        float var = s2 * (1.f / DSTATE) - mu * mu;
        float v = (x - mu) * rsqrtf(var + 1e-5f) * g_state[tid] + be_state[tid];
        sn[tid] = v;
        state_n_out[i * DSTATE + tid] = v;
    }
    __syncthreads();
    const int c = tid;                       // 0..127 output channel
    const float* Wl = W_dist + DRBF * DPAIR;
    const float* Wr = W_dist + (DRBF + DSTATE) * DPAIR;
    float accL = b_dist[c], accR = 0.f;      // fold b_dist into sl
    #pragma unroll
    for (int d = 0; d < DSTATE; ++d) {
        float v = sn[d];
        accL += v * Wl[d * DPAIR + c];
        accR += v * Wr[d * DPAIR + c];
    }
    sl[i * DPAIR + c] = accL;
    sr[i * DPAIR + c] = accR;
}

// ---------------------------------------------------------------------------
// Kernel 2: repack Wd (rows 0..63 of W_dist, f32) into f16 WMMA B-fragments.
// Fragment (t,h): B = Wd[32h..32h+31][16t..16t+15] (32x16).
// Contiguous-k layout used here (A built alike): lane<16 holds col N=lane, K=0..15 in elems 0..15;
//                 lane>=16 holds col N=lane-16, K=16..31.
// ---------------------------------------------------------------------------
__global__ void wd_prep_kernel(const float* __restrict__ W_dist,
                               _Float16* __restrict__ wdh)
{
    const int tid  = threadIdx.x;      // 0..511
    const int frag = tid >> 5;         // t*2 + h, 16 fragments
    const int lane = tid & 31;
    const int t = frag >> 1, h = frag & 1;
    const int hi = lane >> 4;
    const int n  = (t << 4) + (lane & 15);
    #pragma unroll
    for (int e = 0; e < 16; ++e) {
        int k = h * 32 + hi * 16 + e;
        wdh[(frag * 32 + lane) * 16 + e] = (_Float16)W_dist[k * DPAIR + n];
    }
}

// ---------------------------------------------------------------------------
// Kernel 3: msa_out = sctors @ W_sct + b_sct + LN(msa). One block per row.
// ---------------------------------------------------------------------------
__global__ void __launch_bounds__(256) msa_kernel(
    const float* __restrict__ msa, const float* __restrict__ sctors,
    const float* __restrict__ W_sct, const float* __restrict__ b_sct,
    const float* __restrict__ g_msa, const float* __restrict__ be_msa,
    float* __restrict__ msa_out)
{
    __shared__ float red[DMSA];
    __shared__ float sct_s[20];
    const int i = blockIdx.x, c = threadIdx.x;
    if (c < 20) sct_s[c] = sctors[i * 20 + c];
    float x = msa[i * DMSA + c];
    red[c] = x; __syncthreads();
    for (int off = 128; off; off >>= 1) { if (c < off) red[c] += red[c + off]; __syncthreads(); }
    float mu = red[0] * (1.f / DMSA); __syncthreads();
    float dx = x - mu;
    red[c] = dx * dx; __syncthreads();
    for (int off = 128; off; off >>= 1) { if (c < off) red[c] += red[c + off]; __syncthreads(); }
    float inv = rsqrtf(red[0] * (1.f / DMSA) + 1e-5f);
    float s = b_sct[c];
    #pragma unroll
    for (int k = 0; k < 20; ++k) s += sct_s[k] * W_sct[k * DMSA + c];
    msa_out[i * DMSA + c] = s + dx * inv * g_msa[c] + be_msa[c];
}

// ---------------------------------------------------------------------------
// Kernel 4: fused pair kernel.
// Wave = 16 rows (one i, 16 consecutive j). A = rbf feats built in registers,
// 2 WMMAs per 16x16 N-tile (K=64), 8 tiles -> 16 v_wmma per strip.
// B fragments staged once per block in LDS (shared by 8 waves).
// Proj staged in LDS, then fused epilogue: +sl[i] +sr[j] (+b_dist folded)
// + layernorm(pair). pair is streamed with non-temporal load/store so the
// 1 GB stream does not evict the hot ~1MB (sl/sr/Wd/gamma/beta) from L2.
// ---------------------------------------------------------------------------
__global__ void __launch_bounds__(256) pair_kernel(
    const float* __restrict__ pair, const float* __restrict__ xyz,
    const float* __restrict__ g_pair, const float* __restrict__ be_pair,
    const float* __restrict__ sl, const float* __restrict__ sr,
    const _Float16* __restrict__ wdh, float* __restrict__ pair_out)
{
    __shared__ float    proj[8 * 16 * DPAIR];   // 64 KB, 8 KB per wave
    __shared__ _Float16 bsh[16 * 32 * 16];      // 16 KB of B fragments

    // cooperative copy of the 16 B-fragments into LDS (1024 x uint4)
    {
        const uint4* src = (const uint4*)wdh;
        uint4*       dst = (uint4*)bsh;
        #pragma unroll
        for (int q = 0; q < 4; ++q)
            dst[threadIdx.x + q * 256] = src[threadIdx.x + q * 256];
    }

    const int lane  = threadIdx.x & 31;
    const int wave  = threadIdx.x >> 5;
    const int strip = blockIdx.x * 8 + wave;  // 65536 strips total
    const int i  = strip >> 6;
    const int j0 = (strip & 63) << 4;
    const int m  = lane & 15;
    const int hi = lane >> 4;
    const int j  = j0 + m;

    // distance for this lane's row (ca = xyz[:, :, 1, :])
    float dx = xyz[i * 9 + 3] - xyz[j * 9 + 3];
    float dy = xyz[i * 9 + 4] - xyz[j * 9 + 4];
    float dz = xyz[i * 9 + 5] - xyz[j * 9 + 5];
    float D = sqrtf(fmaxf(dx * dx + dy * dy + dz * dz, 1e-12f));

    // A fragments: 16x32 f16 layout -> lane<16: K 0-15, lane>=16: K 16-31
    const float mu0 = 2.0f, step = 20.0f / 63.0f, inv_sigma = 64.0f / 20.0f;
    v16h a0, a1;
    #pragma unroll
    for (int e = 0; e < 16; ++e) {
        int k0 = hi * 16 + e;
        float t0 = (D - (mu0 + (float)k0 * step)) * inv_sigma;
        float t1 = (D - (mu0 + (float)(k0 + 32) * step)) * inv_sigma;
        a0[e] = (_Float16)__expf(-t0 * t0);
        a1[e] = (_Float16)__expf(-t1 * t1);
    }

    __syncthreads();   // B fragments resident in LDS

    const v16h* wdv = (const v16h*)bsh;
    float* myp = &proj[wave * (16 * DPAIR)];
    #pragma unroll
    for (int t = 0; t < 8; ++t) {
        v16h b0 = wdv[(2 * t + 0) * 32 + lane];
        v16h b1 = wdv[(2 * t + 1) * 32 + lane];
        v8f  c  = {};
        c = __builtin_amdgcn_wmma_f32_16x16x32_f16(false, a0, false, b0, (short)0, c, false, false);
        c = __builtin_amdgcn_wmma_f32_16x16x32_f16(false, a1, false, b1, (short)0, c, false, false);
        int nb = (t << 4) + (lane & 15);
        #pragma unroll
        for (int r = 0; r < 8; ++r)            // C layout: VGPR r -> row r (lanes<16) / r+8
            myp[(r + hi * 8) * DPAIR + nb] = c[r];
    }
    __syncthreads();

    // Epilogue: lane handles row m, channel half [hi*64, hi*64+64)
    const size_t row = (size_t)i * LSEQ + (size_t)(j0 + m);
    const f4* prow = (const f4*)(pair + row * DPAIR + hi * 64);
    f4 pr[16];
    float s = 0.f, s2 = 0.f;
    #pragma unroll
    for (int q = 0; q < 16; ++q) {
        f4 v = __builtin_nontemporal_load(prow + q);   // stream pair (TH=NT)
        pr[q] = v;
        s  += v.x + v.y + v.z + v.w;
        s2 += v.x * v.x + v.y * v.y + v.z * v.z + v.w * v.w;
    }
    s  += __shfl_xor(s,  16, 32);
    s2 += __shfl_xor(s2, 16, 32);
    float mu  = s * (1.f / DPAIR);
    float inv = rsqrtf(s2 * (1.f / DPAIR) - mu * mu + 1e-5f);

    const float* pl  = myp + m * DPAIR + hi * 64;
    const float* slr = sl + i * DPAIR + hi * 64;
    const float* srr = sr + (j0 + m) * DPAIR + hi * 64;
    const float* gp  = g_pair  + hi * 64;
    const float* bp  = be_pair + hi * 64;
    f4* orow = (f4*)(pair_out + row * DPAIR + hi * 64);
    #pragma unroll
    for (int q = 0; q < 16; ++q) {
        f4 v = pr[q], o;
        int c0 = q * 4;
        o.x = pl[c0+0] + slr[c0+0] + srr[c0+0] + (v.x - mu) * inv * gp[c0+0] + bp[c0+0];
        o.y = pl[c0+1] + slr[c0+1] + srr[c0+1] + (v.y - mu) * inv * gp[c0+1] + bp[c0+1];
        o.z = pl[c0+2] + slr[c0+2] + srr[c0+2] + (v.z - mu) * inv * gp[c0+2] + bp[c0+2];
        o.w = pl[c0+3] + slr[c0+3] + srr[c0+3] + (v.w - mu) * inv * gp[c0+3] + bp[c0+3];
        __builtin_nontemporal_store(o, orow + q);      // stream pair_out (TH=NT)
    }
}

// ---------------------------------------------------------------------------
extern "C" void kernel_launch(void* const* d_in, const int* in_sizes, int n_in,
                              void* d_out, int out_size, void* d_ws, size_t ws_size,
                              hipStream_t stream)
{
    const float* msa     = (const float*)d_in[0];
    const float* pair    = (const float*)d_in[1];
    const float* xyz     = (const float*)d_in[2];
    const float* state   = (const float*)d_in[3];
    const float* sctors  = (const float*)d_in[4];
    const float* W_dist  = (const float*)d_in[5];
    const float* b_dist  = (const float*)d_in[6];
    const float* g_pair  = (const float*)d_in[7];
    const float* be_pair = (const float*)d_in[8];
    const float* W_sct   = (const float*)d_in[9];
    const float* b_sct   = (const float*)d_in[10];
    const float* g_msa   = (const float*)d_in[11];
    const float* be_msa  = (const float*)d_in[12];
    const float* g_state = (const float*)d_in[13];
    const float* be_state= (const float*)d_in[14];

    float* out       = (float*)d_out;
    float* msa_out   = out;                                        // 1024*256
    float* pair_out  = out + (size_t)LSEQ * DMSA;                  // 1024*1024*128
    float* state_out = pair_out + (size_t)LSEQ * LSEQ * DPAIR;     // 1024*32

    float*    sl  = (float*)d_ws;                   // 512 KB
    float*    sr  = sl + LSEQ * DPAIR;              // 512 KB
    _Float16* wdh = (_Float16*)(sr + LSEQ * DPAIR); // 16 KB of f16 B-fragments

    state_prep_kernel<<<LSEQ, 128, 0, stream>>>(state, g_state, be_state,
                                                W_dist, b_dist, state_out, sl, sr);
    wd_prep_kernel<<<1, 512, 0, stream>>>(W_dist, wdh);
    msa_kernel<<<LSEQ, 256, 0, stream>>>(msa, sctors, W_sct, b_sct,
                                         g_msa, be_msa, msa_out);
    pair_kernel<<<(LSEQ * LSEQ / 16) / 8, 256, 0, stream>>>(
        pair, xyz, g_pair, be_pair, sl, sr, wdh, pair_out);
}
